// AggAtt_68667937129008
// MI455X (gfx1250) — matrix operand order repaired and checked by reading, not compiled, not run on hardware
//
#include <hip/hip_runtime.h>

typedef __attribute__((ext_vector_type(16))) __bf16 v16bf;
typedef __attribute__((ext_vector_type(8)))  __bf16 v8bf;
typedef __attribute__((ext_vector_type(8)))  float  v8f;

#define Bn   2
#define Cc   256
#define Hh   128
#define Ww   128
#define HW   (Hh*Ww)      /* 16384 */
#define Oo   256
#define NTAP 5            /* taps k=0,2,4,6,8 -- odd taps have mask==0 */

/* ---------------- om = conv3x3(inp, w_om)+b_om, 7 channels ----------------- */
__global__ void k_om(const float* __restrict__ inp, const float* __restrict__ w_om,
                     const float* __restrict__ b_om, float* __restrict__ om) {
  int i = blockIdx.x * blockDim.x + threadIdx.x;
  if (i >= Bn * 7 * HW) return;
  int x = i & (Ww - 1);
  int y = (i >> 7) & (Hh - 1);
  int bf = i >> 14;            /* b*7+f */
  int f = bf % 7;
  int b = bf / 7;
  float acc = b_om[f];
  const float* ib = inp + (size_t)b * Cc * HW;
  const float* wb = w_om + (size_t)f * Cc * 9;
  for (int c = 0; c < Cc; ++c) {
    const float* ic = ib + (size_t)c * HW;
    const float* wc = wb + c * 9;
#pragma unroll
    for (int dy = 0; dy < 3; ++dy) {
      int yy = y + dy - 1;
      if (yy < 0 || yy >= Hh) continue;
#pragma unroll
      for (int dx = 0; dx < 3; ++dx) {
        int xx = x + dx - 1;
        if (xx < 0 || xx >= Ww) continue;
        acc += ic[yy * Ww + xx] * wc[dy * 3 + dx];
      }
    }
  }
  om[i] = acc;
}

/* --------- per-pixel/tap bilinear params: 4 premultiplied weights + y0,x0 -- */
__global__ void k_params(const float* __restrict__ om, const float* __restrict__ wh,
                         float* __restrict__ prm) {
  int p = blockIdx.x * blockDim.x + threadIdx.x;
  if (p >= Bn * HW) return;
  int rem = p & (HW - 1);
  int b = p >> 14;
  int y = rem >> 7;
  int x = rem & 127;
  const float* omb = om + (size_t)b * 7 * HW + rem;
  float o0 = omb[0 * HW], o1 = omb[1 * HW];
  float m0 = 1.f / (1.f + __expf(-omb[2 * HW]));
  float m1 = 1.f / (1.f + __expf(-omb[3 * HW]));
  float m2 = 1.f / (1.f + __expf(-omb[4 * HW]));
  float m3 = 1.f / (1.f + __expf(-omb[5 * HW]));
  float m4 = 1.f / (1.f + __expf(-omb[6 * HW]));
  float w = wh[(size_t)b * 2 * HW + rem];
  float h = wh[(size_t)b * 2 * HW + HW + rem];
  /* base-offset cancellation: sample at (y+ry, x+rx) */
  float ry[NTAP] = { -h * 0.5f, -h * 0.5f, o0,  h * 0.5f, 0.f };
  float rx[NTAP] = { -w * 0.5f,  w * 0.5f, o1, -w * 0.5f, w * 0.5f };
  float mm[NTAP] = { m0, m1, m2, m3, m4 };
#pragma unroll
  for (int t = 0; t < NTAP; ++t) {
    float ys = (float)y + ry[t];
    float xs = (float)x + rx[t];
    float y0f = floorf(ys), x0f = floorf(xs);
    float wy1 = ys - y0f, wx1 = xs - x0f;
    float wy0 = 1.f - wy1, wx0 = 1.f - wx1;
    float y1f = y0f + 1.f, x1f = x0f + 1.f;
    bool vy0 = (y0f >= 0.f) && (y0f <= 127.f);
    bool vy1 = (y1f >= 0.f) && (y1f <= 127.f);
    bool vx0 = (x0f >= 0.f) && (x0f <= 127.f);
    bool vx1 = (x1f >= 0.f) && (x1f <= 127.f);
    float m = mm[t];
    float* q = prm + ((size_t)p * NTAP + t) * 6;
    q[0] = (vy0 && vx0) ? wy0 * wx0 * m : 0.f;
    q[1] = (vy0 && vx1) ? wy0 * wx1 * m : 0.f;
    q[2] = (vy1 && vx0) ? wy1 * wx0 * m : 0.f;
    q[3] = (vy1 && vx1) ? wy1 * wx1 * m : 0.f;
    ((int*)q)[4] = (int)y0f;
    ((int*)q)[5] = (int)x0f;
  }
}

/* ---------------- NCHW f32 -> NHWC bf16 ----------------------------------- */
__global__ void k_nhwc(const float* __restrict__ inp, __bf16* __restrict__ nhwc) {
  int i = blockIdx.x * blockDim.x + threadIdx.x;
  if (i >= Bn * Cc * HW) return;
  int rem = i & (HW - 1);
  int c = (i >> 14) & 255;
  int b = i >> 22;
  nhwc[((size_t)b * HW + rem) * Cc + c] = (__bf16)inp[i];
}

/* ------------- weight (O,C,3,3) -> bpack[t][o][c] bf16, k=2t --------------- */
__global__ void k_wpack(const float* __restrict__ wgt, __bf16* __restrict__ bp) {
  int i = blockIdx.x * blockDim.x + threadIdx.x;
  if (i >= NTAP * Oo * Cc) return;
  int c = i & 255;
  int o = (i >> 8) & 255;
  int t = i >> 16;
  bp[i] = (__bf16)wgt[((size_t)o * Cc + c) * 9 + 2 * t];
}

/* async memory->LDS stage of one 256(o) x 32(k) bf16 weight block (16 KB),
   using the CDNA5 GLOBAL_LOAD_ASYNC_TO_LDS_B128 path (ASYNCcnt-tracked). */
__device__ __forceinline__ void stage_async(const __bf16* gsrc, int cb,
                                            unsigned ldsbase, int tid) {
#pragma unroll
  for (int it = 0; it < 8; ++it) {
    int qq = tid + it * 128;               /* 0..1023 chunks of 16B */
    int ol = qq >> 2, cp = qq & 3;
    unsigned long long ga =
        (unsigned long long)(const void*)(gsrc + (size_t)ol * Cc + cb * 32) +
        (unsigned long long)(cp * 16);
    unsigned la = ldsbase + (unsigned)(qq * 16);
    asm volatile("global_load_async_to_lds_b128 %0, %1, off"
                 :: "v"(la), "v"(ga) : "memory");
  }
}

/* ------------- fused gather-GEMM (bf16 WMMA) + bias/relu/1x1 ---------------
   One workgroup = 4 waves = 64 pixels x FULL N=256. Weight blocks are DMA'd
   to a double-buffered LDS region with async-to-LDS so HBM fetch overlaps the
   16 WMMAs per K-step; B fragments are software-pipelined out of LDS. */
__global__ __launch_bounds__(128) void k_dcn(
    const __bf16* __restrict__ nhwc, const __bf16* __restrict__ bp,
    const float* __restrict__ prm, const float* __restrict__ bias,
    const float* __restrict__ w1, const float* __restrict__ b1,
    float* __restrict__ out) {
  __shared__ __align__(16) unsigned char ldsB[2 * 16384]; /* double buffer */
  const int tid = threadIdx.x;
  const int lane = tid & 31;
  const int wv = tid >> 5;
  const int hi = lane >> 4;
  const int nl = lane & 15;
  const int pw = blockIdx.x * 64 + wv * 16;   /* wave's first pixel */

  const int px = pw + nl;                     /* gather pixel for this lane */
  const int b = px >> 14;
  const __bf16* nb = nhwc + (size_t)b * HW * Cc;
  const unsigned lds0 = (unsigned)(size_t)(void*)&ldsB[0]; /* group-seg offset */

  v8f acc[16];
#pragma unroll
  for (int j = 0; j < 16; ++j) acc[j] = (v8f){};

  /* prime buffer 0 with (t=0, cb=0) */
  stage_async(bp, 0, lds0, tid);

  for (int t = 0; t < NTAP; ++t) {
    const float* q = prm + ((size_t)px * NTAP + t) * 6;
    float W00 = q[0], W01 = q[1], W10 = q[2], W11 = q[3];
    int y0 = ((const int*)q)[4];
    int x0 = ((const int*)q)[5];
    int yc0 = min(max(y0, 0), Hh - 1);
    int yc1 = min(max(y0 + 1, 0), Hh - 1);
    int xc0 = min(max(x0, 0), Ww - 1);
    int xc1 = min(max(x0 + 1, 0), Ww - 1);
    const __bf16* p00 = nb + ((yc0 << 7) + xc0) * Cc;
    const __bf16* p01 = nb + ((yc0 << 7) + xc1) * Cc;
    const __bf16* p10 = nb + ((yc1 << 7) + xc0) * Cc;
    const __bf16* p11 = nb + ((yc1 << 7) + xc1) * Cc;

    for (int cb = 0; cb < 8; ++cb) {
      const int step = t * 8 + cb;
      const unsigned curbase = lds0 + (unsigned)((step & 1) * 16384);

      /* publish current buffer: drain our async fills, then workgroup barrier */
      asm volatile("s_wait_asynccnt 0x0" ::: "memory");
      __syncthreads();

      /* kick DMA for the next K-step into the other buffer */
      if (step + 1 < NTAP * 8) {
        int ns = step + 1;
        stage_async(bp + (size_t)(ns >> 3) * Oo * Cc, ns & 7,
                    lds0 + (unsigned)((ns & 1) * 16384), tid);
      }

      /* A fragment: contiguous K layout -> lanes<16 need K{0..15} */
      const int coff = cb * 32 + (hi << 4);
      v8bf c00a = *(const v8bf*)(p00 + coff);
      v8bf c01a = *(const v8bf*)(p01 + coff);
      v8bf c10a = *(const v8bf*)(p10 + coff);
      v8bf c11a = *(const v8bf*)(p11 + coff);
      v8bf c00b = *(const v8bf*)(p00 + coff + 8);
      v8bf c01b = *(const v8bf*)(p01 + coff + 8);
      v8bf c10b = *(const v8bf*)(p10 + coff + 8);
      v8bf c11b = *(const v8bf*)(p11 + coff + 8);
      v16bf af;
#pragma unroll
      for (int i2 = 0; i2 < 8; ++i2) {
        float lo = W00 * (float)c00a[i2] + W01 * (float)c01a[i2] +
                   W10 * (float)c10a[i2] + W11 * (float)c11a[i2];
        float hh = W00 * (float)c00b[i2] + W01 * (float)c01b[i2] +
                   W10 * (float)c10b[i2] + W11 * (float)c11b[i2];
        af[i2] = (__bf16)lo;
        af[i2 + 8] = (__bf16)hh;
      }

      /* B fragments: lane holds N=nl of tile j, K half 'hi' -> 32B in LDS.
         Software-pipeline: preload j+1 before WMMA j. */
      const unsigned char* lb =
          (const unsigned char*)(size_t)(curbase + (unsigned)(nl * 64 + hi * 32));
      const unsigned char* lbg = ldsB + ((step & 1) * 16384 + nl * 64 + hi * 32);
      v16bf bfc = *(const v16bf*)(lbg);
      (void)lb;
#pragma unroll
      for (int j = 0; j < 16; ++j) {
        v16bf bfn;
        if (j < 15) bfn = *(const v16bf*)(lbg + (j + 1) * 16 * 64);
        acc[j] = __builtin_amdgcn_wmma_f32_16x16x32_bf16(
            false, af, false, bfc, (short)0, acc[j], false, false);
        bfc = bfn;
      }
    }
  }

  /* epilogue: +bias, relu, 1x1 (F=2); full o-range is in this wave, so the
     16-lane xor reduction completes each pixel -> plain stores, no atomics */
  float pa[8], pb[8];
#pragma unroll
  for (int r = 0; r < 8; ++r) { pa[r] = 0.f; pb[r] = 0.f; }
#pragma unroll
  for (int j = 0; j < 16; ++j) {
    int o = j * 16 + nl;
    float bo = bias[o];
    float wA = w1[o];
    float wB = w1[Oo + o];
#pragma unroll
    for (int r = 0; r < 8; ++r) {
      float v = acc[j][r] + bo;
      v = v > 0.f ? v : 0.f;
      pa[r] += wA * v;
      pb[r] += wB * v;
    }
  }
#pragma unroll
  for (int r = 0; r < 8; ++r) {
#pragma unroll
    for (int off = 8; off >= 1; off >>= 1) {
      pa[r] += __shfl_xor(pa[r], off, 16);
      pb[r] += __shfl_xor(pb[r], off, 16);
    }
  }
  if (nl == 0) {
    float b1f0 = b1[0], b1f1 = b1[1];
#pragma unroll
    for (int r = 0; r < 8; ++r) {
      int pxo = pw + r + 8 * hi;          /* C/D layout: row r, +8 for hi half */
      int bb = pxo >> 14;
      int rr = pxo & (HW - 1);
      out[(size_t)(bb * 2 + 0) * HW + rr] = pa[r] + b1f0;
      out[(size_t)(bb * 2 + 1) * HW + rr] = pb[r] + b1f1;
    }
  }
}

extern "C" void kernel_launch(void* const* d_in, const int* in_sizes, int n_in,
                              void* d_out, int out_size, void* d_ws, size_t ws_size,
                              hipStream_t stream) {
  const float* inp    = (const float*)d_in[0];
  const float* wh     = (const float*)d_in[1];
  const float* w_om   = (const float*)d_in[2];
  const float* b_om   = (const float*)d_in[3];
  const float* weight = (const float*)d_in[4];
  const float* bias   = (const float*)d_in[5];
  const float* w_1x1  = (const float*)d_in[6];
  const float* b_1x1  = (const float*)d_in[7];
  float* out = (float*)d_out;

  char* ws = (char*)d_ws;
  float*  om   = (float*)(ws);                       /* 2*7*16384*4   = 917504   */
  float*  prm  = (float*)(ws + 917504);              /* 32768*5*6*4   = 3932160  */
  __bf16* nhwc = (__bf16*)(ws + 4849664);            /* 2*16384*256*2 = 16777216 */
  __bf16* bp   = (__bf16*)(ws + 21626880);           /* 5*256*256*2   = 655360   */

  k_om    <<<(Bn * 7 * HW) / 256, 256, 0, stream>>>(inp, w_om, b_om, om);
  k_params<<<(Bn * HW) / 256,     256, 0, stream>>>(om, wh, prm);
  k_nhwc  <<<(Bn * Cc * HW) / 256,256, 0, stream>>>(inp, nhwc);
  k_wpack <<<(NTAP * Oo * Cc) / 256, 256, 0, stream>>>(weight, bp);

  k_dcn<<<512, 128, 0, stream>>>(nhwc, bp, prm, bias, w_1x1, b_1x1, out);
}
